// RingAttention_57715770523894
// MI455X (gfx1250) — matrix operand order repaired and checked by reading, not compiled, not run on hardware
//
#include <hip/hip_runtime.h>
#include <hip/hip_bf16.h>

// ---------------- problem constants ----------------
constexpr int Hd   = 2048;   // hidden
constexpr int NHd  = 16;     // heads
constexpr int HDd  = 128;    // head dim
constexpr int Bd   = 2;      // batch
constexpr int Sd   = 2048;   // seq
constexpr int Md   = Bd * Sd;  // 4096 rows

typedef __attribute__((ext_vector_type(16))) __bf16       v16bf;
typedef __attribute__((ext_vector_type(8)))  float        v8f;
typedef __attribute__((ext_vector_type(4)))  unsigned int v4u;
typedef unsigned short u16;

union BFrag { v16bf bf; v4u u[2]; };

__device__ inline u16 f2bf(float f) {
  union { float f; unsigned u; } c; c.f = f;
  unsigned u = c.u;
  u += 0x7fffu + ((u >> 16) & 1u);   // round-to-nearest-even
  return (u16)(u >> 16);
}

// A-fragment (16x32 bf16, ISA 7.12.2): lane half hl holds K = 8*hl..8*hl+7 and 16+8*hl..+7
__device__ inline v16bf load_a(const u16* row, int hl) {
  BFrag f;
  f.u[0] = *(const v4u*)(row + hl * 8);
  f.u[1] = *(const v4u*)(row + hl * 8 + 16);
  return f.bf;
}
// B-fragment (32x16 bf16): lane half hl holds K = 8*hl..8*hl+7 and 16+8*hl..+7 (matches A)
__device__ inline v16bf load_b(const u16* row, int hl) {
  BFrag f;
  const u16* p = row + hl * 8;
  f.u[0] = *(const v4u*)(p);
  f.u[1] = *(const v4u*)(p + 16);
  return f.bf;
}

__device__ inline v8f wmma_bf(v16bf a, v16bf b, v8f c) {
  return __builtin_amdgcn_wmma_f32_16x16x32_bf16(false, a, false, b, (short)0, c, false, false);
}

// 16B global -> LDS async copy (per-lane), tracked by ASYNCcnt (ISA 08)
__device__ inline void async_cp16(void* lds, const void* g) {
  unsigned l = (unsigned)(size_t)lds;   // low 32 bits of generic shared ptr = LDS byte address
  asm volatile("global_load_async_to_lds_b128 %0, %1, off"
               :: "v"(l), "v"(g) : "memory");
}
__device__ inline void wait_async0() {
  asm volatile("s_wait_asynccnt 0x0" ::: "memory");
}

// ---------------- fp32 -> bf16 convert ----------------
__global__ void cvt_f32_bf16(const float* __restrict__ s, u16* __restrict__ d, int n) {
  int i = blockIdx.x * 256 + threadIdx.x;
  if (i < n) d[i] = f2bf(s[i]);
}

// ---------------- GEMM: C[m,n] = sum_k A[m,k] * W[n,k] ----------------
// block tile 64(M) x 256(N), 8 waves of 32x64; K staged 64 per LDS buffer,
// double-buffered via async-to-LDS copies.
// mode 0: Q -> [b,h,s,d]  mode 1: K -> [b,h,s,d]  mode 2: V -> [b,h,d,s]  mode 3: f32 out
__global__ __launch_bounds__(256)
void gemm_bf16_kernel(const u16* __restrict__ A,
                      const u16* __restrict__ w0, const u16* __restrict__ w1,
                      const u16* __restrict__ w2, const u16* __restrict__ w3,
                      u16* __restrict__ qb, u16* __restrict__ kb, u16* __restrict__ vtb,
                      float* __restrict__ out, int proj)
{
  __shared__ __align__(16) u16 As[2][64][72];    // 64 cols + 8 pad
  __shared__ __align__(16) u16 Bs[2][256][72];

  const int tid  = threadIdx.x;
  const int lane = tid & 31, wv = tid >> 5;
  const int r = lane & 15, hl = lane >> 4;
  const int wm = (wv >> 2) * 32, wn = (wv & 3) * 64;
  const int bm = blockIdx.x * 64, bn = blockIdx.y * 256;
  const int mode = proj ? 3 : (int)blockIdx.z;
  const u16* W = (mode == 0) ? w0 : (mode == 1) ? w1 : (mode == 2) ? w2 : w3;

  v8f z = {};
  v8f acc[2][4];
  #pragma unroll
  for (int i = 0; i < 2; ++i)
    #pragma unroll
    for (int j = 0; j < 4; ++j) acc[i][j] = z;

  const int arow = tid >> 2;           // 0..63
  const int acol = (tid & 3) * 16;     // 0,16,32,48

  auto stage = [&](int s, int buf) {
    const int k0 = s * 64;
    const u16* ap = A + (size_t)(bm + arow) * Hd + k0 + acol;
    async_cp16(&As[buf][arow][acol],     ap);
    async_cp16(&As[buf][arow][acol + 8], ap + 8);
    const u16* bp = W + (size_t)(bn + tid) * Hd + k0;   // thread t stages full W row t
    #pragma unroll
    for (int c = 0; c < 8; ++c)
      async_cp16(&Bs[buf][tid][c * 8], bp + c * 8);
  };

  constexpr int NSTAGE = Hd / 64;   // 32
  stage(0, 0);
  for (int s = 0; s < NSTAGE; ++s) {
    const int buf = s & 1;
    wait_async0();
    __syncthreads();                 // buf full, buf^1 free (all waves done reading it)
    if (s + 1 < NSTAGE) stage(s + 1, buf ^ 1);
    #pragma unroll
    for (int ks = 0; ks < 2; ++ks) {
      v16bf a0 = load_a(&As[buf][wm + r][ks * 32], hl);
      v16bf a1 = load_a(&As[buf][wm + 16 + r][ks * 32], hl);
      v16bf bfr[4];
      #pragma unroll
      for (int j = 0; j < 4; ++j)
        bfr[j] = load_b(&Bs[buf][wn + j * 16 + r][ks * 32], hl);
      #pragma unroll
      for (int j = 0; j < 4; ++j) {
        acc[0][j] = wmma_bf(a0, bfr[j], acc[0][j]);
        acc[1][j] = wmma_bf(a1, bfr[j], acc[1][j]);
      }
    }
  }

  // C layout: lane 0-15 -> N=lane, M=v ; lane 16-31 -> N=lane-16, M=v+8
  #pragma unroll
  for (int i = 0; i < 2; ++i)
    #pragma unroll
    for (int j = 0; j < 4; ++j)
      #pragma unroll
      for (int v = 0; v < 8; ++v) {
        int m = bm + wm + i * 16 + v + 8 * hl;
        int n = bn + wn + j * 16 + r;
        float val = acc[i][j][v];
        if (mode == 3) {
          out[(size_t)m * Hd + n] = val;
        } else {
          int b = m >> 11, s = m & (Sd - 1);
          int h = n >> 7,  d = n & (HDd - 1);
          u16 bvv = f2bf(val);
          if (mode == 0)      qb[((size_t)(b * NHd + h) * Sd + s) * HDd + d] = bvv;
          else if (mode == 1) kb[((size_t)(b * NHd + h) * Sd + s) * HDd + d] = bvv;
          else                vtb[((size_t)(b * NHd + h) * HDd + d) * Sd + s] = bvv;
        }
      }
}

// ---------------- flash attention, causal ----------------
// grid: (S/64 q-tiles, B*NH). block = 128 threads = 4 waves; wave owns 16 q rows x 128 d.
// K / V^T blocks double-buffered in LDS via async-to-LDS copies.
__global__ __launch_bounds__(128)
void flash_attn_kernel(const u16* __restrict__ qb, const u16* __restrict__ kb,
                       const u16* __restrict__ vtb, u16* __restrict__ attnb)
{
  __shared__ __align__(16) u16 Ks[2][64][136];   // K block [s][d], padded
  __shared__ __align__(16) u16 Vts[2][128][72];  // V^T block [d][s], padded
  __shared__ __align__(16) u16 Ps[4][16][72];    // per-wave P tile [row][col]

  const int tid  = threadIdx.x;
  const int lane = tid & 31, wv = tid >> 5;
  const int r = lane & 15, hl = lane >> 4;
  const int bh = blockIdx.y;
  const int qt = blockIdx.x;
  const int qr0 = qt * 64 + wv * 16;

  // resident Q fragments (16 rows x hd=128 -> 4 A frags)
  const u16* qrow = qb + ((size_t)bh * Sd + qr0 + r) * HDd;
  v16bf aq[4];
  #pragma unroll
  for (int kk = 0; kk < 4; ++kk) aq[kk] = load_a(qrow + kk * 32, hl);

  v8f z = {};
  v8f oacc[8];
  #pragma unroll
  for (int dt = 0; dt < 8; ++dt) oacc[dt] = z;
  float mrow[8], lrow[8];
  #pragma unroll
  for (int v = 0; v < 8; ++v) { mrow[v] = -INFINITY; lrow[v] = 0.f; }

  const float scale = 0.08838834764831845f;  // 1/sqrt(128)

  auto stage = [&](int jb, int buf) {
    #pragma unroll
    for (int c = 0; c < 8; ++c) {
      int idx = tid * 8 + c;
      int kr = idx >> 4, kc = (idx & 15) * 8;
      async_cp16(&Ks[buf][kr][kc],
                 kb + ((size_t)bh * Sd + jb * 64 + kr) * HDd + kc);
      int vr = idx >> 3, vc = (idx & 7) * 8;
      async_cp16(&Vts[buf][vr][vc],
                 vtb + ((size_t)bh * HDd + vr) * Sd + jb * 64 + vc);
    }
  };

  stage(0, 0);
  for (int jb = 0; jb <= qt; ++jb) {
    const int buf = jb & 1;
    wait_async0();
    __syncthreads();                 // buf full, buf^1 free
    if (jb + 1 <= qt) stage(jb + 1, buf ^ 1);

    // scores: S = Q K^T  (16 WMMAs)
    v8f sc[4];
    sc[0] = z; sc[1] = z; sc[2] = z; sc[3] = z;
    #pragma unroll
    for (int nt = 0; nt < 4; ++nt)
      #pragma unroll
      for (int kk = 0; kk < 4; ++kk) {
        v16bf bk = load_b(&Ks[buf][nt * 16 + r][kk * 32], hl);
        sc[nt] = wmma_bf(aq[kk], bk, sc[nt]);
      }

    // scale + causal mask
    #pragma unroll
    for (int nt = 0; nt < 4; ++nt) {
      int coln = jb * 64 + nt * 16 + r;
      #pragma unroll
      for (int v = 0; v < 8; ++v) {
        int qg = qr0 + v + 8 * hl;
        float x = sc[nt][v] * scale;
        if (coln > qg) x = -INFINITY;
        sc[nt][v] = x;
      }
    }

    // streaming softmax stats (rows live across the 16-lane halves of wave32)
    #pragma unroll
    for (int v = 0; v < 8; ++v) {
      float t = fmaxf(fmaxf(sc[0][v], sc[1][v]), fmaxf(sc[2][v], sc[3][v]));
      #pragma unroll
      for (int mk = 1; mk < 16; mk <<= 1) t = fmaxf(t, __shfl_xor(t, mk, 16));
      float mn = fmaxf(mrow[v], t);
      float corr = __expf(mrow[v] - mn);
      mrow[v] = mn;
      lrow[v] *= corr;
      #pragma unroll
      for (int dt = 0; dt < 8; ++dt) oacc[dt][v] *= corr;
      float rs = 0.f;
      #pragma unroll
      for (int nt = 0; nt < 4; ++nt) {
        float p = __expf(sc[nt][v] - mn);
        sc[nt][v] = p;
        rs += p;
      }
      #pragma unroll
      for (int mk = 1; mk < 16; mk <<= 1) rs += __shfl_xor(rs, mk, 16);
      lrow[v] += rs;
    }

    // P: C-layout -> A-fragment reshape via (wave-private) LDS
    #pragma unroll
    for (int nt = 0; nt < 4; ++nt)
      #pragma unroll
      for (int v = 0; v < 8; ++v)
        Ps[wv][v + 8 * hl][nt * 16 + r] = f2bf(sc[nt][v]);
    __builtin_amdgcn_wave_barrier();                 // wave-internal reshape only
    asm volatile("s_wait_dscnt 0x0" ::: "memory");   // P stores visible to own wave

    // O += P V  (16 WMMAs; B frags contiguous thanks to V^T layout)
    #pragma unroll
    for (int kk = 0; kk < 2; ++kk) {
      v16bf ap = load_a(&Ps[wv][r][kk * 32], hl);
      #pragma unroll
      for (int dt = 0; dt < 8; ++dt) {
        v16bf bvf = load_b(&Vts[buf][dt * 16 + r][kk * 32], hl);
        oacc[dt] = wmma_bf(ap, bvf, oacc[dt]);
      }
    }
  }

  // epilogue: attn[b, s, h*hd + d] bf16 (row-major for the output-proj GEMM)
  int b = bh >> 4, h = bh & 15;
  #pragma unroll
  for (int dt = 0; dt < 8; ++dt)
    #pragma unroll
    for (int v = 0; v < 8; ++v) {
      int qg = qr0 + v + 8 * hl;
      int d = dt * 16 + r;
      float o = oacc[dt][v] / (lrow[v] + 1e-8f);
      attnb[(size_t)(b * Sd + qg) * Hd + h * HDd + d] = f2bf(o);
    }
}

// ---------------- host-side launch ----------------
extern "C" void kernel_launch(void* const* d_in, const int* in_sizes, int n_in,
                              void* d_out, int out_size, void* d_ws, size_t ws_size,
                              hipStream_t stream)
{
  const float* x  = (const float*)d_in[0];
  const float* wq = (const float*)d_in[1];
  const float* wk = (const float*)d_in[2];
  const float* wv = (const float*)d_in[3];
  const float* wo = (const float*)d_in[4];

  char* ws = (char*)d_ws;
  const size_t MB = 1024 * 1024;
  u16* xb    = (u16*)(ws);             // 16 MB
  u16* wqb   = (u16*)(ws + 16 * MB);   //  8 MB each
  u16* wkb   = (u16*)(ws + 24 * MB);
  u16* wvb   = (u16*)(ws + 32 * MB);
  u16* wob   = (u16*)(ws + 40 * MB);
  u16* qb    = (u16*)(ws + 48 * MB);   // 16 MB each
  u16* kb    = (u16*)(ws + 64 * MB);
  u16* vtb   = (u16*)(ws + 80 * MB);
  u16* attnb = (u16*)(ws + 96 * MB);   // total 112 MB

  const int xN = Md * Hd;
  const int wN = Hd * Hd;
  cvt_f32_bf16<<<(xN + 255) / 256, 256, 0, stream>>>(x,  xb,  xN);
  cvt_f32_bf16<<<(wN + 255) / 256, 256, 0, stream>>>(wq, wqb, wN);
  cvt_f32_bf16<<<(wN + 255) / 256, 256, 0, stream>>>(wk, wkb, wN);
  cvt_f32_bf16<<<(wN + 255) / 256, 256, 0, stream>>>(wv, wvb, wN);
  cvt_f32_bf16<<<(wN + 255) / 256, 256, 0, stream>>>(wo, wob, wN);

  dim3 gq(Md / 64, Hd / 256, 3);
  gemm_bf16_kernel<<<gq, 256, 0, stream>>>(xb, wqb, wkb, wvb, wob, qb, kb, vtb, nullptr, 0);

  dim3 ga(Sd / 64, Bd * NHd, 1);
  flash_attn_kernel<<<ga, 128, 0, stream>>>(qb, kb, vtb, attnb);

  dim3 gp(Md / 64, Hd / 256, 1);
  gemm_bf16_kernel<<<gp, 256, 0, stream>>>(attnb, wqb, wkb, wvb, wob, qb, kb, vtb,
                                           (float*)d_out, 1);
}
